// CrossAttention_65051574665792
// MI455X (gfx1250) — compile-verified, matrix operand order checked by reading, not run on hardware
//
#include <hip/hip_runtime.h>
#include <hip/hip_bf16.h>

// ---------------------------------------------------------------------------
// Types for CDNA5 WMMA (wave32): bf16 A/B fragments (16 x bf16 = 8 VGPRs),
// f32 C/D accumulator (8 VGPRs).
// ---------------------------------------------------------------------------
typedef __attribute__((ext_vector_type(16))) __bf16          v16bf;
typedef __attribute__((ext_vector_type(8)))  float           v8f;
typedef __attribute__((ext_vector_type(8)))  unsigned short  v8u16;
typedef __attribute__((ext_vector_type(4)))  unsigned int    v4u;
typedef __attribute__((ext_vector_type(8)))  int             v8i;
typedef __attribute__((ext_vector_type(4)))  int             v4i;

union Frag {
    v16bf bf;
    v8u16 u[2];
};

__device__ __forceinline__ v8f wmma_bf16(const Frag& a, const Frag& b, v8f c) {
    return __builtin_amdgcn_wmma_f32_16x16x32_bf16(
        /*neg_a=*/false, a.bf, /*neg_b=*/false, b.bf,
        /*c_mod=*/(short)0, c, /*reuse_a=*/false, /*reuse_b=*/false);
}

// float -> bf16 (round to nearest even), kept as raw ushort
__device__ __forceinline__ unsigned short f2bf(float f) {
    unsigned int u = __float_as_uint(f);
    u += 0x7FFFu + ((u >> 16) & 1u);
    return (unsigned short)(u >> 16);
}

// ---------------------------------------------------------------------------
// Tensor Data Mover: 2-D tile load global -> LDS, descriptor per ISA 08 §8.
// data_size = 8B units; per-row LDS padding reproduces the bank-conflict-free
// strides. Issued by wave 0 only; completion tracked with TENSORcnt.
// Toolchain here exposes the 6-arg builtin:
//   (uint32x4 g0, int32x8 g1, int32x4 g2, int32x4 g3, int32x8 g4, i32 cpol)
// ---------------------------------------------------------------------------
#if defined(__has_builtin)
# if __has_builtin(__builtin_amdgcn_tensor_load_to_lds)
#  define HAS_TDM 1
# endif
#endif
#ifndef HAS_TDM
# define HAS_TDM 0
#endif

// tile_w_b: row length in bytes (mult of 16); tile_h: rows;
// stride_b: global row stride in bytes (mult of 8);
// pad_int: pad every (8 << pad_int) bytes; pad_amt: (pad_amt+1) DWORDs of pad.
__device__ __forceinline__ void tdm_load_2d(unsigned short* lds_ptr,
                                            const unsigned short* gptr,
                                            unsigned int tile_w_b,
                                            unsigned int tile_h,
                                            unsigned int stride_b,
                                            unsigned int pad_int,
                                            unsigned int pad_amt,
                                            unsigned int lds_stride_e) {
#if HAS_TDM
    (void)lds_stride_e;
    const unsigned long long ga = (unsigned long long)(uintptr_t)gptr;
    const unsigned int lds_off  = (unsigned int)(uintptr_t)lds_ptr; // AS(3) byte offset
    v4u g0;
    g0[0] = 1u;                                              // count=1 (valid D#)
    g0[1] = lds_off;                                         // lds_addr
    g0[2] = (unsigned int)ga;                                // global_addr[31:0]
    g0[3] = (unsigned int)((ga >> 32) & 0x01FFFFFFull)       // global_addr[56:32]
          | (2u << 30);                                      // type=2 ("image")
    const unsigned int td0 = stride_b >> 3;                  // tensor_dim0 (8B units)
    const unsigned int td1 = 1u << 20;                       // tensor_dim1 (ample)
    v8i g1;
    g1[0] = (int)((3u << 16)                                 // data_size = 8B
                | (1u << 20)                                 // pad_enable
                | (pad_int << 22) | (pad_amt << 25));
    g1[1] = (int)((td0 & 0xFFFFu) << 16);                    // tensor_dim0[15:0]
    g1[2] = (int)((td0 >> 16) | ((td1 & 0xFFFFu) << 16));    // td0[31:16] | td1[15:0]
    g1[3] = (int)((td1 >> 16) | ((tile_w_b >> 3) << 16));    // td1[31:16] | tile_dim0
    g1[4] = (int)(tile_h & 0xFFFFu);                         // tile_dim1; tile_dim2=0
    g1[5] = (int)(stride_b >> 3);                            // tensor_dim0_stride lo32
    g1[6] = 0;                                               // stride hi | dim1_stride lo
    g1[7] = 0;
    v4i g2; g2[0] = 0; g2[1] = 0; g2[2] = 0; g2[3] = 0;      // 2-D: groups 2/3 zero
    v4i g3 = g2;
    v8i g4;                                                   // extended block: zero
    g4[0] = 0; g4[1] = 0; g4[2] = 0; g4[3] = 0;
    g4[4] = 0; g4[5] = 0; g4[6] = 0; g4[7] = 0;
    __builtin_amdgcn_tensor_load_to_lds(g0, g1, g2, g3, g4, 0);
#else
    // fallback: lane-parallel cooperative copy by the issuing wave
    const int lane = threadIdx.x & 31;
    for (unsigned int row = lane; row < tile_h; row += 32) {
        const unsigned short* src = gptr + (size_t)row * (stride_b >> 1);
        unsigned short*       dst = lds_ptr + (size_t)row * lds_stride_e;
        for (unsigned int c = 0; c < (tile_w_b >> 1); c += 8)
            *(v8u16*)(dst + c) = *(const v8u16*)(src + c);
    }
    (void)pad_int; (void)pad_amt;
#endif
}

__device__ __forceinline__ void tdm_wait0() {
#if HAS_TDM
    __builtin_amdgcn_s_wait_tensorcnt(0);
#else
    __builtin_amdgcn_s_waitcnt(0);
#endif
}

// ---------------------------------------------------------------------------
// Problem constants
// ---------------------------------------------------------------------------
#define NTOK   4096
#define DMODEL 1024
#define NHEAD  16
#define DHEAD  64
#define QK_SCALE 0.125f   // 64^-0.5

// ---------------------------------------------------------------------------
// Prep kernels: fp32 -> bf16 cast, and fp32 -> bf16 transpose (for weights)
// ---------------------------------------------------------------------------
__global__ __launch_bounds__(256)
void cast_bf16_kernel(const float* __restrict__ in, unsigned short* __restrict__ out, int n) {
    int i = blockIdx.x * 256 + threadIdx.x;
    if (i < n) out[i] = f2bf(in[i]);
}

__global__ __launch_bounds__(256)
void transpose_bf16_kernel(const float* __restrict__ in, unsigned short* __restrict__ out,
                           int rows, int cols) {
    __shared__ float tile[32][33];
    int r0 = blockIdx.y * 32, c0 = blockIdx.x * 32;
    int tx = threadIdx.x, ty = threadIdx.y;
    for (int k = 0; k < 32; k += 8)
        tile[ty + k][tx] = in[(size_t)(r0 + ty + k) * cols + (c0 + tx)];
    __syncthreads();
    for (int k = 0; k < 32; k += 8)
        out[(size_t)(c0 + ty + k) * rows + (r0 + tx)] = f2bf(tile[tx][ty + k]);
}

// ---------------------------------------------------------------------------
// Tiled bf16 GEMM:  C[M x N] = A[M x K] * Bt[N x K]^T   (Bt row-major N x K)
// 256 threads = 8 wave32 -> 128x128 C tile, K-step 32, TDM double-buffered.
// MODE 0: bf16 out, scaled | MODE 1: bf16 out, transposed | MODE 2: fp32+bias
// ---------------------------------------------------------------------------
#define LDA 40   // (32 elems = 64B) + 16B pad : conflict-free ds_load_b128
template<int MODE>
__global__ __launch_bounds__(256)
void gemm_bf16(const unsigned short* __restrict__ A,
               const unsigned short* __restrict__ Bt,
               void* __restrict__ Cout,
               const float* __restrict__ bias,
               int M, int N, int K, float scale) {
    const int tid  = threadIdx.x;
    const int lane = tid & 31, wid = tid >> 5;
    const int wm   = wid >> 1, wn = wid & 1;
    const int m0   = blockIdx.x * 128, n0 = blockIdx.y * 128;
    const int hh   = lane >> 4, r = lane & 15;

    __shared__ __attribute__((aligned(128))) unsigned short lA[2][128 * LDA];
    __shared__ __attribute__((aligned(128))) unsigned short lB[2][128 * LDA];

    v8f acc[2][4] = {};

    // prefetch K-chunk 0 (rows of 64B, pad 16B => pad_int=3 (8<<3=64), pad_amt=3 (4 DW))
    if (wid == 0) {
        tdm_load_2d(&lA[0][0], A  + (size_t)m0 * K, 64, 128, K * 2, 3, 3, LDA);
        tdm_load_2d(&lB[0][0], Bt + (size_t)n0 * K, 64, 128, K * 2, 3, 3, LDA);
        tdm_wait0();
    }
    __syncthreads();

    int buf = 0;
    for (int k0 = 0; k0 < K; k0 += 32, buf ^= 1) {
        if (wid == 0 && (k0 + 32) < K) {
            tdm_load_2d(&lA[buf ^ 1][0], A  + (size_t)m0 * K + k0 + 32, 64, 128, K * 2, 3, 3, LDA);
            tdm_load_2d(&lB[buf ^ 1][0], Bt + (size_t)n0 * K + k0 + 32, 64, 128, K * 2, 3, 3, LDA);
        }
        const unsigned short* cA = &lA[buf][0];
        const unsigned short* cB = &lB[buf][0];

        Frag a[2], b[4];
        #pragma unroll
        for (int mt = 0; mt < 2; ++mt) {
            const int rl = wm * 32 + mt * 16 + r;
            a[mt].u[0] = *(const v8u16*)(cA + rl * LDA + hh * 8);
            a[mt].u[1] = *(const v8u16*)(cA + rl * LDA + 16 + hh * 8);
        }
        #pragma unroll
        for (int nt = 0; nt < 4; ++nt) {
            const int rl = wn * 64 + nt * 16 + r;
            b[nt].u[0] = *(const v8u16*)(cB + rl * LDA + hh * 8);
            b[nt].u[1] = *(const v8u16*)(cB + rl * LDA + 16 + hh * 8);
        }
        #pragma unroll
        for (int mt = 0; mt < 2; ++mt)
            #pragma unroll
            for (int nt = 0; nt < 4; ++nt)
                acc[mt][nt] = wmma_bf16(a[mt], b[nt], acc[mt][nt]);

        if (wid == 0) tdm_wait0();   // next buffer landed
        __syncthreads();             // everyone done reading this buffer
    }

    #pragma unroll
    for (int mt = 0; mt < 2; ++mt)
        #pragma unroll
        for (int nt = 0; nt < 4; ++nt)
            #pragma unroll
            for (int i = 0; i < 8; ++i) {
                const int row = m0 + wm * 32 + mt * 16 + 8 * hh + i;
                const int col = n0 + wn * 64 + nt * 16 + r;
                const float v = acc[mt][nt][i] * scale;
                if (MODE == 0)
                    ((unsigned short*)Cout)[(size_t)row * N + col] = f2bf(v);
                else if (MODE == 1)
                    ((unsigned short*)Cout)[(size_t)col * M + row] = f2bf(v);
                else
                    ((float*)Cout)[(size_t)row * N + col] = v + bias[col];
            }
}

// ---------------------------------------------------------------------------
// Flash attention: grid = (NTOK/128, NHEAD), block = 256 (8 wave32).
// Wave w owns 16 query rows; 64-key chunks (halves per-key softmax overhead),
// K/V tiles streamed by TDM with double buffering. Q pre-scaled by 1/sqrt(dh).
//   Qb[token][feat], Kb[token][feat], Vt[feat][token], Ob[token][feat] (bf16)
// ---------------------------------------------------------------------------
#define ALD 72   // (64 elems = 128B) + 16B pad
__global__ __launch_bounds__(256)
void attn_kernel(const unsigned short* __restrict__ Qb,
                 const unsigned short* __restrict__ Kb,
                 const unsigned short* __restrict__ Vt,
                 unsigned short* __restrict__ Ob) {
    const int tid  = threadIdx.x;
    const int lane = tid & 31, wid = tid >> 5;
    const int hh   = lane >> 4, r = lane & 15;
    const int h    = blockIdx.y;
    const int q0   = blockIdx.x * 128 + wid * 16;

    __shared__ __attribute__((aligned(128))) unsigned short lK[2][64 * ALD]; // keys x d
    __shared__ __attribute__((aligned(128))) unsigned short lV[2][64 * ALD]; // d x keys
    __shared__ __attribute__((aligned(128))) unsigned short lP[8][16 * ALD]; // P staging

    // Q fragments, resident for the whole key sweep
    Frag aq[2];
    {
        const unsigned short* qrow = Qb + (size_t)(q0 + r) * DMODEL + h * DHEAD;
        #pragma unroll
        for (int c = 0; c < 2; ++c) {
            aq[c].u[0] = *(const v8u16*)(qrow + c * 32 + hh * 8);
            aq[c].u[1] = *(const v8u16*)(qrow + c * 32 + 16 + hh * 8);
        }
    }

    v8f o[4] = {};
    float mrow[8], lrow[8];
    #pragma unroll
    for (int i = 0; i < 8; ++i) { mrow[i] = -1e30f; lrow[i] = 0.0f; }

    // prefetch chunk 0 (rows of 128B, pad 16B => pad_int=4 (8<<4=128), pad_amt=3)
    if (wid == 0) {
        tdm_load_2d(&lK[0][0], Kb + (size_t)0 * DMODEL + h * DHEAD,
                    128, 64, DMODEL * 2, 4, 3, ALD);
        tdm_load_2d(&lV[0][0], Vt + (size_t)(h * DHEAD) * NTOK,
                    128, 64, NTOK * 2, 4, 3, ALD);
        tdm_wait0();
    }
    __syncthreads();

    int buf = 0;
    for (int j0 = 0; j0 < NTOK; j0 += 64, buf ^= 1) {
        if (wid == 0 && (j0 + 64) < NTOK) {
            tdm_load_2d(&lK[buf ^ 1][0], Kb + (size_t)(j0 + 64) * DMODEL + h * DHEAD,
                        128, 64, DMODEL * 2, 4, 3, ALD);
            tdm_load_2d(&lV[buf ^ 1][0], Vt + (size_t)(h * DHEAD) * NTOK + (j0 + 64),
                        128, 64, NTOK * 2, 4, 3, ALD);
        }
        const unsigned short* K_ = &lK[buf][0];
        const unsigned short* V_ = &lV[buf][0];

        // S = Q*K^T : 16 q-rows x 64 keys = four 16x16 tiles, K-dim 64 (2 chunks)
        v8f s[4] = {};
        #pragma unroll
        for (int nt = 0; nt < 4; ++nt)
            #pragma unroll
            for (int c = 0; c < 2; ++c) {
                Frag bk;
                const int rl = nt * 16 + r;
                bk.u[0] = *(const v8u16*)(K_ + rl * ALD + c * 32 + hh * 8);
                bk.u[1] = *(const v8u16*)(K_ + rl * ALD + c * 32 + 16 + hh * 8);
                s[nt] = wmma_bf16(aq[c], bk, s[nt]);
            }

        // online softmax; lane's row i = q-row (i + 8*hh); xor masks 1,2,4,8
        // reduce across the 16-lane half of the wave32 C layout
        #pragma unroll
        for (int i = 0; i < 8; ++i) {
            float x = fmaxf(fmaxf(s[0][i], s[1][i]), fmaxf(s[2][i], s[3][i]));
            x = fmaxf(x, __shfl_xor(x, 1, 32));
            x = fmaxf(x, __shfl_xor(x, 2, 32));
            x = fmaxf(x, __shfl_xor(x, 4, 32));
            x = fmaxf(x, __shfl_xor(x, 8, 32));
            const float mnew  = fmaxf(mrow[i], x);
            const float alpha = __expf(mrow[i] - mnew);
            mrow[i] = mnew;
            float rs = 0.0f;
            #pragma unroll
            for (int nt = 0; nt < 4; ++nt) {
                const float p = __expf(s[nt][i] - mnew);
                s[nt][i] = p;
                rs += p;
            }
            rs += __shfl_xor(rs, 1, 32);
            rs += __shfl_xor(rs, 2, 32);
            rs += __shfl_xor(rs, 4, 32);
            rs += __shfl_xor(rs, 8, 32);
            lrow[i] = lrow[i] * alpha + rs;
            #pragma unroll
            for (int dt = 0; dt < 4; ++dt) o[dt][i] *= alpha;
        }

        // re-layout P (C-layout -> A-layout) through per-wave LDS staging
        unsigned short* pw = &lP[wid][0];
        #pragma unroll
        for (int i = 0; i < 8; ++i) {
            const int rr = i + 8 * hh;
            #pragma unroll
            for (int nt = 0; nt < 4; ++nt)
                pw[rr * ALD + nt * 16 + r] = f2bf(s[nt][i]);
        }
        // intra-wave LDS RAW hazard: CDNA5 split DS counter wait
        __builtin_amdgcn_wave_barrier();
        asm volatile("s_wait_dscnt 0" ::: "memory");
        __builtin_amdgcn_wave_barrier();

        Frag pa[2];
        #pragma unroll
        for (int c = 0; c < 2; ++c) {
            pa[c].u[0] = *(const v8u16*)(pw + r * ALD + c * 32 + hh * 8);
            pa[c].u[1] = *(const v8u16*)(pw + r * ALD + c * 32 + 16 + hh * 8);
        }

        // O += P*V : 16 q-rows x 64 d, K-dim = 64 keys (2 chunks)
        #pragma unroll
        for (int dt = 0; dt < 4; ++dt) {
            const int rl = dt * 16 + r;
            #pragma unroll
            for (int c = 0; c < 2; ++c) {
                Frag bv;
                bv.u[0] = *(const v8u16*)(V_ + rl * ALD + c * 32 + hh * 8);
                bv.u[1] = *(const v8u16*)(V_ + rl * ALD + c * 32 + 16 + hh * 8);
                o[dt] = wmma_bf16(pa[c], bv, o[dt]);
            }
        }

        if (wid == 0) tdm_wait0();   // next K/V tiles landed
        __syncthreads();             // all waves done with this buffer
    }

    #pragma unroll
    for (int dt = 0; dt < 4; ++dt)
        #pragma unroll
        for (int i = 0; i < 8; ++i) {
            const int row = q0 + i + 8 * hh;
            const int col = h * DHEAD + dt * 16 + r;
            Ob[(size_t)row * DMODEL + col] = f2bf(o[dt][i] / lrow[i]);
        }
}

// ---------------------------------------------------------------------------
// Host: cast -> QKV GEMMs -> flash attention -> output GEMM
// ---------------------------------------------------------------------------
extern "C" void kernel_launch(void* const* d_in, const int* in_sizes, int n_in,
                              void* d_out, int out_size, void* d_ws, size_t ws_size,
                              hipStream_t stream) {
    const float* x  = (const float*)d_in[0];
    const float* Wq = (const float*)d_in[1];
    const float* Wk = (const float*)d_in[2];
    const float* Wv = (const float*)d_in[3];
    const float* Wo = (const float*)d_in[4];
    const float* bo = (const float*)d_in[5];
    float* out = (float*)d_out;

    char* ws = (char*)d_ws;
    const size_t SZ_X = (size_t)NTOK * DMODEL * sizeof(unsigned short);   // 8 MB
    const size_t SZ_W = (size_t)DMODEL * DMODEL * sizeof(unsigned short); // 2 MB
    unsigned short* xb  = (unsigned short*)(ws);             ws += SZ_X;
    unsigned short* WqT = (unsigned short*)(ws);             ws += SZ_W;
    unsigned short* WkT = (unsigned short*)(ws);             ws += SZ_W;
    unsigned short* WvT = (unsigned short*)(ws);             ws += SZ_W;
    unsigned short* WoT = (unsigned short*)(ws);             ws += SZ_W;
    unsigned short* Qb  = (unsigned short*)(ws);             ws += SZ_X;
    unsigned short* Kb  = (unsigned short*)(ws);             ws += SZ_X;
    unsigned short* Vt  = (unsigned short*)(ws);             ws += SZ_X;
    unsigned short* Ob  = (unsigned short*)(ws);             ws += SZ_X;

    {
        const int n = NTOK * DMODEL;
        cast_bf16_kernel<<<n / 256, 256, 0, stream>>>(x, xb, n);
        dim3 tb(32, 8), tg(DMODEL / 32, DMODEL / 32);
        transpose_bf16_kernel<<<tg, tb, 0, stream>>>(Wq, WqT, DMODEL, DMODEL);
        transpose_bf16_kernel<<<tg, tb, 0, stream>>>(Wk, WkT, DMODEL, DMODEL);
        transpose_bf16_kernel<<<tg, tb, 0, stream>>>(Wv, WvT, DMODEL, DMODEL);
        transpose_bf16_kernel<<<tg, tb, 0, stream>>>(Wo, WoT, DMODEL, DMODEL);
    }
    {
        dim3 g(NTOK / 128, DMODEL / 128);
        gemm_bf16<0><<<g, 256, 0, stream>>>(xb, WqT, Qb, nullptr,
                                            NTOK, DMODEL, DMODEL, QK_SCALE);
        gemm_bf16<0><<<g, 256, 0, stream>>>(xb, WkT, Kb, nullptr,
                                            NTOK, DMODEL, DMODEL, 1.0f);
        gemm_bf16<1><<<g, 256, 0, stream>>>(xb, WvT, Vt, nullptr,
                                            NTOK, DMODEL, DMODEL, 1.0f);
    }
    {
        dim3 g(NTOK / 128, NHEAD);
        attn_kernel<<<g, 256, 0, stream>>>(Qb, Kb, Vt, Ob);
    }
    {
        dim3 g(NTOK / 128, DMODEL / 128);
        gemm_bf16<2><<<g, 256, 0, stream>>>(Ob, WoT, out, bo,
                                            NTOK, DMODEL, DMODEL, 1.0f);
    }
}
